// LSTM_52501680226341
// MI455X (gfx1250) — matrix operand order repaired and checked by reading, not compiled, not run on hardware
//
#include <hip/hip_runtime.h>
#include <math.h>

typedef _Float16 half_t;
typedef __attribute__((ext_vector_type(16))) _Float16 v16h;
typedef __attribute__((ext_vector_type(8)))  _Float16 v8h;
typedef __attribute__((ext_vector_type(8)))  float    v8f;

#define INPUT_DIM 166
#define KPAD      192
#define HID       64
#define G4        256      // 4*HID
#define BATCH     256
#define TSTEPS    2000
#define MT        80       // time-tile for input projection (5 x 16)

__device__ inline v8f wmma_f16(v16h a, v16h b, v8f c) {
  return __builtin_amdgcn_wmma_f32_16x16x32_f16(false, a, false, b, (short)0, c, false, false);
}

__device__ inline v16h join8(v8h lo, v8h hi) {
  return __builtin_shufflevector(lo, hi, 0,1,2,3,4,5,6,7,8,9,10,11,12,13,14,15);
}

#if defined(__has_builtin)
#if __has_builtin(__builtin_amdgcn_rcpf)
#define FAST_RCP(x) __builtin_amdgcn_rcpf(x)
#endif
#if __has_builtin(__builtin_amdgcn_exp2f)
#define FAST_EXP2(x) __builtin_amdgcn_exp2f(x)
#endif
#if __has_builtin(__builtin_amdgcn_tanhf)
#define FAST_TANH(x) __builtin_amdgcn_tanhf(x)
#endif
#endif
#ifndef FAST_RCP
#define FAST_RCP(x) (1.0f / (x))
#endif
#ifndef FAST_EXP2
#define FAST_EXP2(x) exp2f(x)
#endif

__device__ inline float fast_sigmoid(float x) {
  // 1 / (1 + 2^(-x*log2e))
  return FAST_RCP(1.0f + FAST_EXP2(x * -1.4426950408889634f));
}
#ifndef FAST_TANH
__device__ inline float tanh_fallback(float x) {
  x = fminf(fmaxf(x, -15.0f), 15.0f);
  float e = FAST_EXP2(x * 2.8853900817779268f);  // e^(2x)
  return (e - 1.0f) * FAST_RCP(e + 1.0f);
}
#define FAST_TANH(x) tanh_fallback(x)
#endif

// ---------------- Phase 0: convert weights to f16, fuse biases ----------------
__global__ void lstm_prep(const float* __restrict__ W_ih, const float* __restrict__ W_hh,
                          const float* __restrict__ b_ih, const float* __restrict__ b_hh,
                          half_t* __restrict__ wih_h, half_t* __restrict__ whh_h,
                          float* __restrict__ bias) {
  int tid = blockIdx.x * blockDim.x + threadIdx.x;
  int stride = gridDim.x * blockDim.x;
  for (int i = tid; i < G4 * KPAD; i += stride) {
    int n = i / KPAD, k = i - n * KPAD;
    float v = (k < INPUT_DIM) ? W_ih[n * INPUT_DIM + k] : 0.0f;
    wih_h[i] = (half_t)v;
  }
  for (int i = tid; i < G4 * HID; i += stride) whh_h[i] = (half_t)W_hh[i];
  for (int i = tid; i < G4; i += stride) bias[i] = b_ih[i] + b_hh[i];
}

// ---------------- Phase 1: xg[t][b][4H] = x_btc @ W_ih^T + bias (f16 out) ----
// x: [B][INPUT][T]; grid = B * (T/MT); block = 256 (8 waves)
__global__ __launch_bounds__(256) void lstm_input_proj(
    const float* __restrict__ x, const half_t* __restrict__ wih,
    const float* __restrict__ bias, half_t* __restrict__ xg) {
  __shared__ __align__(16) half_t xs[MT * 40];  // [t][c], row pad 40

  const int tid  = threadIdx.x;
  const int lane = tid & 31;
  const int wv   = tid >> 5;         // 0..7
  const int ln   = lane & 15;
  const int hi   = lane >> 4;
  const int b    = blockIdx.x / (TSTEPS / MT);
  const int t0   = (blockIdx.x % (TSTEPS / MT)) * MT;
  const float* xb = x + (size_t)b * INPUT_DIM * TSTEPS;

  v8f acc[5][2] = {};

  for (int kt = 0; kt < 6; ++kt) {
    const int kb = kt * 32;
    __syncthreads();
    // stage x slab [32 c][MT t] into LDS transposed -> xs[t][c], f16
    #pragma unroll
    for (int q = 0; q < (32 * MT) / 256; ++q) {
      int idx = q * 256 + tid;
      int cc = idx / MT, tt = idx - cc * MT;
      float v = 0.0f;
      if (kb + cc < INPUT_DIM) v = xb[(size_t)(kb + cc) * TSTEPS + t0 + tt];
      xs[tt * 40 + cc] = (half_t)v;
    }
    __syncthreads();

    // B fragments for this wave's two 16-col tiles: lane n=ln, K = 16*hi + [0,16)
    v16h Bf[2];
    #pragma unroll
    for (int nt = 0; nt < 2; ++nt) {
      const half_t* wp = wih + (wv * 32 + nt * 16 + ln) * KPAD + kb + 16 * hi;
      Bf[nt] = join8(*(const v8h*)wp, *(const v8h*)(wp + 8));
    }

    #pragma unroll
    for (int mb = 0; mb < 5; ++mb) {
      // A fragment: row m = mb*16+ln, K = 16*hi + [0,16)
      const half_t* ap = xs + (mb * 16 + ln) * 40 + 16 * hi;
      v16h Af = join8(*(const v8h*)ap, *(const v8h*)(ap + 8));
      acc[mb][0] = wmma_f16(Af, Bf[0], acc[mb][0]);
      acc[mb][1] = wmma_f16(Af, Bf[1], acc[mb][1]);
    }
  }

  const float bz0 = bias[wv * 32 + ln];
  const float bz1 = bias[wv * 32 + 16 + ln];
  #pragma unroll
  for (int mb = 0; mb < 5; ++mb) {
    #pragma unroll
    for (int r = 0; r < 8; ++r) {
      int tg = t0 + mb * 16 + hi * 8 + r;            // C layout: lanes16-31 -> M+8
      size_t base = ((size_t)tg * BATCH + b) * G4;
      xg[base + wv * 32 + ln]      = (half_t)(acc[mb][0][r] + bz0);
      xg[base + wv * 32 + 16 + ln] = (half_t)(acc[mb][1][r] + bz1);
    }
  }
}

// ---------------- Phase 2: sequential recurrence + final FC ------------------
// grid = B/16 workgroups (independent batch tiles), block = 128 (4 waves)
// Wave w owns hidden columns [16w, 16w+16) in all FOUR gates (N-tiles w, w+4,
// w+8, w+12) so the i/f/g/o values for each cell live in one lane's registers:
// no LDS gate exchange, c stays in registers, one barrier per step.
// h is double-buffered in LDS: read buf[t&1], write buf[1-(t&1)].
__global__ __launch_bounds__(128) void lstm_recurrent(
    const half_t* __restrict__ xg, const half_t* __restrict__ whh,
    const float* __restrict__ fc_w, const float* __restrict__ fc_b,
    float* __restrict__ out) {
  __shared__ __align__(16) half_t hbuf[2][16 * HID];  // h tile, f16, double-buffered

  const int tid  = threadIdx.x;
  const int lane = tid & 31;
  const int wv   = tid >> 5;         // 0..3
  const int ln   = lane & 15;
  const int hi   = lane >> 4;
  const int b0   = blockIdx.x * 16;

  // Persistent W_hh B-fragments: gate g -> N cols [g*64 + wv*16, +16), K {0,32}
  v16h Bf[4][2];
  #pragma unroll
  for (int g = 0; g < 4; ++g) {
    #pragma unroll
    for (int kt = 0; kt < 2; ++kt) {
      const half_t* wp = whh + (g * 64 + wv * 16 + ln) * HID + kt * 32 + 16 * hi;
      Bf[g][kt] = join8(*(const v8h*)wp, *(const v8h*)(wp + 8));
    }
  }

  for (int i = tid; i < 16 * HID; i += 128) hbuf[0][i] = (half_t)0.0f;
  float creg[8] = {};

  // software pipeline: xg fragment for step t lives in registers before step t
  half_t xn[4][8];
  {
    const half_t* p = xg + (size_t)b0 * G4;
    #pragma unroll
    for (int g = 0; g < 4; ++g)
      #pragma unroll
      for (int r = 0; r < 8; ++r)
        xn[g][r] = p[(hi * 8 + r) * G4 + g * 64 + wv * 16 + ln];
  }
  __syncthreads();

  for (int t = 0; t < TSTEPS; ++t) {
    const int rb = t & 1, wb = rb ^ 1;

    // consume pipelined xg -> f32 accumulators (C layout: r -> M=8*hi+r, N=ln)
    v8f acc[4];
    #pragma unroll
    for (int g = 0; g < 4; ++g)
      #pragma unroll
      for (int r = 0; r < 8; ++r)
        acc[g][r] = (float)xn[g][r];

    // issue next step's xg loads; latency hidden behind this whole step
    if (t + 1 < TSTEPS) {
      const half_t* p = xg + ((size_t)(t + 1) * BATCH + b0) * G4;
      #pragma unroll
      for (int g = 0; g < 4; ++g)
        #pragma unroll
        for (int r = 0; r < 8; ++r)
          xn[g][r] = p[(hi * 8 + r) * G4 + g * 64 + wv * 16 + ln];
    }

    // A fragments from h_t (LDS read buffer): row m = ln
    const half_t* hp = hbuf[rb] + ln * HID;
    v16h A0 = join8(*(const v8h*)(hp + 16 * hi),      *(const v8h*)(hp + 8 + 16 * hi));
    v16h A1 = join8(*(const v8h*)(hp + 32 + 16 * hi), *(const v8h*)(hp + 40 + 16 * hi));

    #pragma unroll
    for (int g = 0; g < 4; ++g) {
      acc[g] = wmma_f16(A0, Bf[g][0], acc[g]);
      acc[g] = wmma_f16(A1, Bf[g][1], acc[g]);
    }

    // gates entirely in registers: acc[0]=i, acc[1]=f, acc[2]=g, acc[3]=o
    #pragma unroll
    for (int r = 0; r < 8; ++r) {
      float iv = fast_sigmoid(acc[0][r]);
      float fv = fast_sigmoid(acc[1][r]);
      float gv = FAST_TANH(acc[2][r]);
      float ov = fast_sigmoid(acc[3][r]);
      float c = fv * creg[r] + iv * gv;
      creg[r] = c;
      hbuf[wb][(hi * 8 + r) * HID + wv * 16 + ln] = (half_t)(ov * FAST_TANH(c));
    }
    __syncthreads();
  }

  // final FC: h_T is in hbuf[TSTEPS & 1]
  const half_t* hT = hbuf[TSTEPS & 1];
  if (tid < 16 * 7) {
    int mm = tid / 7, cls = tid - mm * 7;
    float s = fc_b[cls];
    for (int k = 0; k < HID; ++k)
      s += (float)hT[mm * HID + k] * fc_w[cls * HID + k];
    out[(b0 + mm) * 7 + cls] = s;
  }
}

extern "C" void kernel_launch(void* const* d_in, const int* in_sizes, int n_in,
                              void* d_out, int out_size, void* d_ws, size_t ws_size,
                              hipStream_t stream) {
  const float* x    = (const float*)d_in[0];
  const float* W_ih = (const float*)d_in[1];
  const float* W_hh = (const float*)d_in[2];
  const float* b_ih = (const float*)d_in[3];
  const float* b_hh = (const float*)d_in[4];
  const float* fc_w = (const float*)d_in[5];
  const float* fc_b = (const float*)d_in[6];
  float* out = (float*)d_out;

  char* ws = (char*)d_ws;
  half_t* wih_h = (half_t*)ws;                               // 256*192*2 = 98304
  half_t* whh_h = (half_t*)(ws + 98304);                     // 256*64*2  = 32768
  float*  bias  = (float*)(ws + 98304 + 32768);              // 256*4     = 1024
  half_t* xg    = (half_t*)(ws + 98304 + 32768 + 1024);      // T*B*256*2 = 262144000

  lstm_prep<<<64, 256, 0, stream>>>(W_ih, W_hh, b_ih, b_hh, wih_h, whh_h, bias);
  lstm_input_proj<<<BATCH * (TSTEPS / MT), 256, 0, stream>>>(x, wih_h, bias, xg);
  lstm_recurrent<<<BATCH / 16, 128, 0, stream>>>(xg, whh_h, fc_w, fc_b, out);
}
